// TopKSelector_90761248899103
// MI455X (gfx1250) — matrix operand order repaired and checked by reading, not compiled, not run on hardware
//
#include <hip/hip_runtime.h>
#include <hip/hip_bf16.h>

// ---------------------------------------------------------------------------
// TopK selector: LayerNorm -> (x @ W1 + b1) -> GELU -> (@ W2 + b2) -> top-k
// -> gather.  GEMM1 on v_wmma_f32_16x16x32_f16.
// ROWS=64 per block amortizes W1 (B-operand) L2 traffic 4x vs ROWS=16:
// 1024 blocks x 512KB = 512MB of L2 reads, no longer the bottleneck vs the
// 285MB / ~12us HBM floor (23.3 TB/s).
// ---------------------------------------------------------------------------

#define B_   8
#define L_   8192
#define D_   1024
#define H_   256
#define K_   512
#define ROWS 64          // rows (b,l) per block in scorer kernel
#define THREADS 512      // 16 waves; wave w owns N-tile w (16 cols)
#define MTILES 4         // 4 M-tiles of 16 rows per wave
#define XH_STRIDE 1032   // 1024 + 8 halves pad -> spreads LDS banks

typedef __attribute__((ext_vector_type(16))) _Float16 v16h;
typedef __attribute__((ext_vector_type(8)))  float    v8f;
typedef __attribute__((ext_vector_type(4)))  float    f32x4;

union FragH { v16h h; f32x4 q[2]; };

// ---------------------------------------------------------------------------
// Kernel 0: W1 [D,H] f32  ->  w1t [H,D] f16  (B-operand friendly layout)
// ---------------------------------------------------------------------------
__global__ void prep_w1_kernel(const float* __restrict__ W1,
                               _Float16* __restrict__ w1t) {
    const int h = blockIdx.x;                       // 0..255
    for (int d = threadIdx.x; d < D_; d += blockDim.x)
        w1t[h * D_ + d] = (_Float16)W1[d * H_ + h];
}

// ---------------------------------------------------------------------------
// Kernel 1: fused LayerNorm + MLP scorer.
//   512 threads (16 waves), 64 rows per block, grid = B*L/64 = 1024.
//   Wave w: LN for rows 4w..4w+3, then WMMA: N-tile w  x  M-tiles 0..3.
// ---------------------------------------------------------------------------
__global__ __launch_bounds__(THREADS)
void scorer_kernel(const float* __restrict__ feat,
                   const float* __restrict__ gamma,
                   const float* __restrict__ beta,
                   const _Float16* __restrict__ w1t,
                   const float* __restrict__ b1,
                   const float* __restrict__ w2,
                   const float* __restrict__ b2,
                   float* __restrict__ scores) {
    __shared__ __align__(16) _Float16 xh[ROWS * XH_STRIDE]; // ~129 KB
    __shared__ float score_lds[ROWS];

    const int tid  = threadIdx.x;
    const int wave = tid >> 5;       // 0..15  == N-tile id
    const int lane = tid & 31;
    const int hi   = lane >> 4;      // half-wave id (0/1)
    const int n    = lane & 15;      // column / row-within-tile id

    const size_t rowBase = (size_t)blockIdx.x * ROWS;
    const float* fbase   = feat + rowBase * D_;

    if (tid < ROWS) score_lds[tid] = b2[0];   // + SCORER_BIAS (== 0)

    // ---- LayerNorm: wave w handles rows 4w..4w+3 ----
    #pragma unroll
    for (int rr = 0; rr < 4; ++rr) {
        const int r = wave * 4 + rr;
        const float* fr = fbase + (size_t)r * D_;
        float sum = 0.f, sq = 0.f;
        for (int k = lane; k < D_; k += 32) {
            float v = fr[k];
            sum += v; sq += v * v;
        }
        #pragma unroll
        for (int m = 16; m > 0; m >>= 1) {
            sum += __shfl_xor(sum, m, 32);
            sq  += __shfl_xor(sq,  m, 32);
        }
        const float mean = sum * (1.0f / D_);
        const float var  = sq * (1.0f / D_) - mean * mean;
        const float rs   = rsqrtf(var + 1e-5f);
        for (int k = lane; k < D_; k += 32) {
            float v = (fr[k] - mean) * rs * gamma[k] + beta[k];
            xh[r * XH_STRIDE + k] = (_Float16)v;
        }
    }
    __syncthreads();

    // ---- GEMM: [64 x 1024] x [1024 x 16-per-wave] ----
    v8f acc[MTILES];
    #pragma unroll
    for (int m = 0; m < MTILES; ++m) acc[m] = (v8f){};

    // A-operand (16x32 f16): lane holds row M = lane&15 of its M-tile;
    // halves 0-7 at K = hi*8, halves 8-15 at K = 16 + hi*8.
    const _Float16* ap[MTILES];
    #pragma unroll
    for (int m = 0; m < MTILES; ++m)
        ap[m] = &xh[(m * 16 + n) * XH_STRIDE + hi * 8];

    // B-operand (32x16 f16): lane holds col N = lane&15;
    // halves 0-7 at K = hi*8, halves 8-15 at K = 16 + hi*8.
    const _Float16* bp = w1t + (wave * 16 + n) * D_ + hi * 8;

    for (int ks = 0; ks < D_ / 32; ++ks) {
        FragH fb;
        fb.q[0] = *(const f32x4*)(bp + ks * 32);
        fb.q[1] = *(const f32x4*)(bp + ks * 32 + 16);
        #pragma unroll
        for (int m = 0; m < MTILES; ++m) {
            FragH fa;
            fa.q[0] = *(const f32x4*)(ap[m] + ks * 32);
            fa.q[1] = *(const f32x4*)(ap[m] + ks * 32 + 16);
            acc[m] = __builtin_amdgcn_wmma_f32_16x16x32_f16(
                         false, fa.h, false, fb.h, (short)0, acc[m],
                         false, false);
        }
    }

    // ---- Epilogue: +b1, exact GELU, * W2, reduce into per-row score ----
    const int h     = wave * 16 + n;
    const float b1v = b1[h];
    const float w2v = w2[h];
    #pragma unroll
    for (int m = 0; m < MTILES; ++m) {
        #pragma unroll
        for (int v = 0; v < 8; ++v) {
            // C layout: VGPR v holds row M = m*16 + v + 8*hi, col N = n.
            float x = acc[m][v] + b1v;
            float g = 0.5f * x * (1.0f + erff(x * 0.70710678118654752f));
            float p = g * w2v;
            #pragma unroll
            for (int mm = 1; mm < 16; mm <<= 1) p += __shfl_xor(p, mm, 32);
            if (n == 0) atomicAdd(&score_lds[m * 16 + v + 8 * hi], p);
        }
    }
    __syncthreads();
    if (tid < ROWS) scores[rowBase + tid] = score_lds[tid];
}

// ---------------------------------------------------------------------------
// Kernel 2: per-batch top-512 of 8192 via full bitonic sort in LDS.
// Key = (orderable(score) << 32) | (8191 - idx)  -> descending sort gives
// largest score first, ties broken by smallest index (torch.topk semantics).
// ---------------------------------------------------------------------------
__global__ __launch_bounds__(1024)
void topk_kernel(const float* __restrict__ scores, int* __restrict__ idx_out) {
    __shared__ unsigned long long keys[L_];
    const int b = blockIdx.x;
    const float* s = scores + (size_t)b * L_;

    for (int i = threadIdx.x; i < L_; i += 1024) {
        unsigned u = __float_as_uint(s[i]);
        u = (u & 0x80000000u) ? ~u : (u | 0x80000000u);
        keys[i] = ((unsigned long long)u << 32) | (unsigned)(L_ - 1 - i);
    }
    for (int size = 2; size <= L_; size <<= 1) {
        for (int stride = size >> 1; stride > 0; stride >>= 1) {
            __syncthreads();
            for (int t = threadIdx.x; t < L_ / 2; t += 1024) {
                const int lo = 2 * t - (t & (stride - 1));
                const int hx = lo + stride;
                unsigned long long a = keys[lo], c = keys[hx];
                const bool desc = ((lo & size) == 0);
                if (desc ? (a < c) : (a > c)) { keys[lo] = c; keys[hx] = a; }
            }
        }
    }
    __syncthreads();
    for (int j = threadIdx.x; j < K_; j += 1024)
        idx_out[b * K_ + j] = (L_ - 1) - (int)(keys[j] & 0xffffffffu);
}

// ---------------------------------------------------------------------------
// Kernel 3: gather original (un-normalized) feature rows of the top-k.
// ---------------------------------------------------------------------------
__global__ __launch_bounds__(256)
void gather_kernel(const float* __restrict__ feat, const int* __restrict__ idx,
                   float* __restrict__ selected) {
    const int j = blockIdx.x;   // 0..511
    const int b = blockIdx.y;   // 0..7
    const int id = idx[b * K_ + j];
    const f32x4* src = (const f32x4*)(feat + ((size_t)b * L_ + id) * D_);
    f32x4* dst = (f32x4*)(selected + ((size_t)b * K_ + j) * D_);
    dst[threadIdx.x] = src[threadIdx.x];     // 256 lanes x 16B = 1024 floats
}

// ---------------------------------------------------------------------------
extern "C" void kernel_launch(void* const* d_in, const int* in_sizes, int n_in,
                              void* d_out, int out_size, void* d_ws, size_t ws_size,
                              hipStream_t stream) {
    const float* feat  = (const float*)d_in[0];
    const float* gamma = (const float*)d_in[1];
    const float* beta  = (const float*)d_in[2];
    const float* W1    = (const float*)d_in[3];
    const float* b1    = (const float*)d_in[4];
    const float* W2    = (const float*)d_in[5];
    const float* b2    = (const float*)d_in[6];
    (void)in_sizes; (void)n_in; (void)out_size; (void)ws_size;

    float* out      = (float*)d_out;
    float* selected = out;                                       // [8,512,1024]
    float* scores   = out + (size_t)B_ * K_ * D_;                // [8,8192]
    int*   idx      = (int*)(out + (size_t)B_ * K_ * D_ + (size_t)B_ * L_); // [8,512]

    _Float16* w1t = (_Float16*)d_ws;                             // 512 KB scratch

    prep_w1_kernel<<<H_, 256, 0, stream>>>(W1, w1t);
    scorer_kernel<<<(B_ * L_) / ROWS, THREADS, 0, stream>>>(
        feat, gamma, beta, w1t, b1, W2, b2, scores);
    topk_kernel<<<B_, 1024, 0, stream>>>(scores, idx);
    gather_kernel<<<dim3(K_, B_), 256, 0, stream>>>(feat, idx, selected);
}
